// GraphConvNet_40140764348830
// MI455X (gfx1250) — compile-verified, matrix operand order checked by reading, not run on hardware
//
#include <hip/hip_runtime.h>

typedef _Float16 v8h  __attribute__((ext_vector_type(8)));
typedef _Float16 v16h __attribute__((ext_vector_type(16)));
typedef float    v8f  __attribute__((ext_vector_type(8)));

#define NNODES 8192
#define DIN    2000
#define DPAD   2048
#define H0     512
#define H1     256
#define H2     128
#define NBINS  4096
// idx = round(0.99*(8192*8192-1)) = 66437774 ; need cumulative count >= idx+1
#define QTARGET 66437775ull

enum { F_ABS = 1, F_RELU = 2, F_LOADACC = 4, F_STOREACC = 8, F_ST16 = 16 };

// ---------------------------------------------------------------------------
// elementwise / prep kernels
// ---------------------------------------------------------------------------

__global__ void rownorm_kernel(const float* __restrict__ x, _Float16* __restrict__ xn,
                               int D, int Dp) {
  const int row = blockIdx.x;
  __shared__ float red[256];
  float s = 0.f;
  for (int c = threadIdx.x; c < D; c += 256) {
    float v = x[(size_t)row * D + c];
    s += v * v;
  }
  red[threadIdx.x] = s;
  __syncthreads();
  for (int w = 128; w > 0; w >>= 1) {
    if (threadIdx.x < w) red[threadIdx.x] += red[threadIdx.x + w];
    __syncthreads();
  }
  const float inv = 1.0f / fmaxf(sqrtf(red[0]), 1e-8f);
  for (int c = threadIdx.x; c < Dp; c += 256) {
    float v = (c < D) ? x[(size_t)row * D + c] * inv : 0.f;
    xn[(size_t)row * Dp + c] = (_Float16)v;
  }
}

__global__ void convpad_kernel(const float* __restrict__ src, _Float16* __restrict__ dst,
                               int R, int C, int Cp) {
  const size_t total = (size_t)R * Cp;
  for (size_t i = (size_t)blockIdx.x * 256 + threadIdx.x; i < total;
       i += (size_t)gridDim.x * 256) {
    int r = (int)(i / Cp), c = (int)(i % Cp);
    dst[i] = (c < C) ? (_Float16)src[(size_t)r * C + c] : (_Float16)0.f;
  }
}

__global__ void transpose_f16_kernel(const _Float16* __restrict__ in,
                                     _Float16* __restrict__ out, int R, int C) {
  __shared__ _Float16 tile[32][33];
  const int bx = blockIdx.x * 32;  // column of in
  const int by = blockIdx.y * 32;  // row of in
  const int tx = threadIdx.x, ty = threadIdx.y;
  for (int i = ty; i < 32; i += 8) {
    int r = by + i, c = bx + tx;
    tile[i][tx] = (r < R && c < C) ? in[(size_t)r * C + c] : (_Float16)0.f;
  }
  __syncthreads();
  for (int i = ty; i < 32; i += 8) {
    int r = bx + i, c = by + tx;  // out is C x R
    if (r < C && c < R) out[(size_t)r * R + c] = tile[tx][i];
  }
}

__global__ void zero_bins_kernel(unsigned* bins) {
  int i = blockIdx.x * 256 + threadIdx.x;
  if (i < NBINS) bins[i] = 0u;
}

__global__ void hist_kernel(const _Float16* __restrict__ sim, unsigned* __restrict__ bins) {
  __shared__ unsigned lb[NBINS];
  for (int i = threadIdx.x; i < NBINS; i += 256) lb[i] = 0u;
  __syncthreads();
  const size_t n = (size_t)NNODES * NNODES;
  for (size_t i = (size_t)blockIdx.x * 256 + threadIdx.x; i < n;
       i += (size_t)gridDim.x * 256) {
    float v = (float)sim[i];
    int b = (int)(v * (float)NBINS);
    b = b < 0 ? 0 : (b > NBINS - 1 ? NBINS - 1 : b);
    atomicAdd(&lb[b], 1u);
  }
  __syncthreads();
  for (int i = threadIdx.x; i < NBINS; i += 256)
    if (lb[i]) atomicAdd(&bins[i], lb[i]);
}

__global__ void find_eps_kernel(const unsigned* __restrict__ bins, float* __restrict__ eps) {
  if (threadIdx.x == 0 && blockIdx.x == 0) {
    unsigned long long cum = 0;
    float e = 1.0f;
    for (int b = 0; b < NBINS; ++b) {
      cum += bins[b];
      if (cum >= QTARGET) { e = (float)b / (float)NBINS; break; }
    }
    eps[0] = e;
  }
}

__global__ void thresh_deg_kernel(_Float16* __restrict__ sim, float* __restrict__ inv_deg,
                                  const float* __restrict__ eps_p) {
  const int row = blockIdx.x;
  const float eps = eps_p[0];
  __shared__ unsigned red[256];
  unsigned cnt = 0;
  for (int j = threadIdx.x; j < NNODES; j += 256) {
    size_t idx = (size_t)row * NNODES + j;
    bool a = (float)sim[idx] >= eps;
    sim[idx] = a ? (_Float16)1.f : (_Float16)0.f;
    cnt += a ? 1u : 0u;
  }
  red[threadIdx.x] = cnt;
  __syncthreads();
  for (int w = 128; w > 0; w >>= 1) {
    if (threadIdx.x < w) red[threadIdx.x] += red[threadIdx.x + w];
    __syncthreads();
  }
  if (threadIdx.x == 0) inv_deg[row] = 1.0f / fmaxf((float)red[0], 1.0f);
}

__global__ void head_kernel(const _Float16* __restrict__ h2, const float* __restrict__ W,
                            const float* __restrict__ b, float* __restrict__ out) {
  int r = blockIdx.x * 256 + threadIdx.x;
  if (r >= NNODES) return;
  float s = b[0];
#pragma unroll 4
  for (int c = 0; c < H2; ++c) s += (float)h2[(size_t)r * H2 + c] * W[c];
  out[r] = 1.0f / (1.0f + expf(-s));
}

// ---------------------------------------------------------------------------
// WMMA GEMM: C(MxN) = epilogue( A(MxK) * B(NxK)^T )
// block = 256 threads (8 wave32), block tile 128x64, wave tile 32x32.
// K-panels (depth 32) are staged global->LDS with GLOBAL_LOAD_ASYNC_TO_LDS_B128
// (ASYNCcnt), double buffered; each panel feeds 4 v_wmma_f32_16x16x32_f16 / wave.
// ---------------------------------------------------------------------------

__device__ __forceinline__ void async_cp16(unsigned lds_off, const _Float16* gptr) {
  // CDNA5 async copy: LDS destination offset in VDST vgpr, 64-bit global addr,
  // tracked by ASYNCcnt (CDNA5 ISA).
  unsigned long long ga = (unsigned long long)(uintptr_t)gptr;
  asm volatile("global_load_async_to_lds_b128 %0, %1, off"
               :
               : "v"(lds_off), "v"(ga)
               : "memory");
}

__device__ __forceinline__ void wait_async0() {
  asm volatile("s_wait_asynccnt 0x0" ::: "memory");
}

__device__ __forceinline__ v16h frag_ld(const _Float16* base, int lane) {
  // 16-bit A/B fragment layout (ISA 7.12.2): lane m=lane&15 holds row m;
  // lanes 0-15: K 0..7 then 16..23 ; lanes 16-31: K 8..15 then 24..31.
  const int r = lane & 15;
  const int kb = (lane >> 4) << 3;  // 0 or 8
  const _Float16* p = base + r * 32 + kb;
  v8h lo = *(const v8h*)(p);
  v8h hi = *(const v8h*)(p + 16);
  return __builtin_shufflevector(lo, hi, 0, 1, 2, 3, 4, 5, 6, 7, 8, 9, 10, 11, 12, 13, 14, 15);
}

__device__ __forceinline__ void epi(v8f c, int mg, int ng, int lane,
                                    _Float16* C16, float* Cacc, int ldc,
                                    const float* bias, const float* rowscale, int flags) {
  const int n = ng + (lane & 15);
  const int mb = mg + ((lane >> 4) << 3);  // C layout: vgpr v -> M=v (+8 for hi lanes)
  const float bv = bias ? bias[n] : 0.0f;
#pragma unroll
  for (int v = 0; v < 8; ++v) {
    const int m = mb + v;
    float val = c[v];
    if (flags & F_LOADACC) val += Cacc[(size_t)m * ldc + n];
    val += bv;
    if (flags & F_ABS) val = fabsf(val);
    if (rowscale) val *= rowscale[m];
    if (flags & F_RELU) val = fmaxf(val, 0.0f);
    if (flags & F_STOREACC) Cacc[(size_t)m * ldc + n] = val;
    if (flags & F_ST16) C16[(size_t)m * ldc + n] = (_Float16)val;
  }
}

__global__ __launch_bounds__(256) void wmma_gemm_kernel(
    const _Float16* __restrict__ A, int lda, const _Float16* __restrict__ B, int ldb,
    _Float16* __restrict__ C16, float* __restrict__ Cacc, int ldc,
    const float* __restrict__ bias, const float* __restrict__ rowscale, int K, int flags) {
  __shared__ __align__(16) _Float16 As[2][128 * 32];  // 8 KB per buffer
  __shared__ __align__(16) _Float16 Bs[2][64 * 32];   // 4 KB per buffer

  const int t = threadIdx.x;
  const int m0 = blockIdx.x * 128;
  const int n0 = blockIdx.y * 64;
  const int lane = t & 31;
  const int wave = t >> 5;
  const int wm = wave >> 1;  // 0..3
  const int wn = wave & 1;   // 0..1

  const int arow = t >> 2;       // 0..63
  const int achk = (t & 3) * 8;  // 8-half (16B) chunk within K panel
  const size_t aoff0 = (size_t)(m0 + arow) * lda + achk;
  const size_t aoff1 = (size_t)(m0 + 64 + arow) * lda + achk;
  const size_t boff = (size_t)(n0 + arow) * ldb + achk;

  // LDS byte offsets for this thread's three 16B async stores (per buffer).
  const unsigned as_base = (unsigned)(uintptr_t)(void*)&As[0][0];
  const unsigned bs_base = (unsigned)(uintptr_t)(void*)&Bs[0][0];
  const unsigned a0l = as_base + (unsigned)(arow * 32 + achk) * 2u;
  const unsigned a1l = as_base + (unsigned)((arow + 64) * 32 + achk) * 2u;
  const unsigned b0l = bs_base + (unsigned)(arow * 32 + achk) * 2u;
  const unsigned ASB = 128u * 32u * 2u;  // bytes per As buffer
  const unsigned BSB = 64u * 32u * 2u;   // bytes per Bs buffer

  v8f c00 = {}, c01 = {}, c10 = {}, c11 = {};

  // prologue: async-stage K-panel 0 into buffer 0
  async_cp16(a0l, A + aoff0);
  async_cp16(a1l, A + aoff1);
  async_cp16(b0l, B + boff);
  wait_async0();
  __syncthreads();

  const int KT = K >> 5;
  for (int kt = 0; kt < KT; ++kt) {
    const int cur = kt & 1;
    const bool more = (kt + 1) < KT;
    if (more) {  // async copies for the next panel overlap the WMMAs below
      const unsigned nb = (unsigned)(cur ^ 1);
      const size_t kk = (size_t)(kt + 1) << 5;
      async_cp16(a0l + nb * ASB, A + aoff0 + kk);
      async_cp16(a1l + nb * ASB, A + aoff1 + kk);
      async_cp16(b0l + nb * BSB, B + boff + kk);
      __builtin_prefetch(A + aoff0 + kk + 32, 0, 1);  // global_prefetch_b8
    }

    const _Float16* ab = &As[cur][(wm * 32) * 32];
    const _Float16* bb = &Bs[cur][(wn * 32) * 32];
    v16h a0 = frag_ld(ab, lane);
    v16h a1 = frag_ld(ab + 16 * 32, lane);
    v16h b0 = frag_ld(bb, lane);
    v16h b1 = frag_ld(bb + 16 * 32, lane);

    c00 = __builtin_amdgcn_wmma_f32_16x16x32_f16(false, a0, false, b0, (short)0, c00, false, false);
    c01 = __builtin_amdgcn_wmma_f32_16x16x32_f16(false, a0, false, b1, (short)0, c01, false, false);
    c10 = __builtin_amdgcn_wmma_f32_16x16x32_f16(false, a1, false, b0, (short)0, c10, false, false);
    c11 = __builtin_amdgcn_wmma_f32_16x16x32_f16(false, a1, false, b1, (short)0, c11, false, false);

    if (more) wait_async0();  // next panel resident before the barrier releases
    __syncthreads();
  }

  const int mg = m0 + wm * 32;
  const int ng = n0 + wn * 32;
  epi(c00, mg, ng, lane, C16, Cacc, ldc, bias, rowscale, flags);
  epi(c01, mg, ng + 16, lane, C16, Cacc, ldc, bias, rowscale, flags);
  epi(c10, mg + 16, ng, lane, C16, Cacc, ldc, bias, rowscale, flags);
  epi(c11, mg + 16, ng + 16, lane, C16, Cacc, ldc, bias, rowscale, flags);
}

// ---------------------------------------------------------------------------
// host orchestration
// ---------------------------------------------------------------------------

static inline size_t alignup(size_t x) { return (x + 255) & ~(size_t)255; }

extern "C" void kernel_launch(void* const* d_in, const int* in_sizes, int n_in,
                              void* d_out, int out_size, void* d_ws, size_t ws_size,
                              hipStream_t stream) {
  (void)in_sizes; (void)n_in; (void)out_size; (void)ws_size;
  const float* x = (const float*)d_in[0];
  const float* W_in = (const float*)d_in[1];
  const float* Wl1 = (const float*)d_in[2];
  const float* bl1 = (const float*)d_in[3];
  const float* Wr1 = (const float*)d_in[4];
  const float* Wl2 = (const float*)d_in[5];
  const float* bl2 = (const float*)d_in[6];
  const float* Wr2 = (const float*)d_in[7];
  const float* W_out = (const float*)d_in[8];
  const float* b_out = (const float*)d_in[9];
  float* out = (float*)d_out;

  // workspace carving
  char* ws = (char*)d_ws;
  size_t off = 0;
  auto carve = [&](size_t bytes) { void* p = ws + off; off += alignup(bytes); return p; };
  _Float16* xn_h   = (_Float16*)carve((size_t)NNODES * DPAD * 2);
  _Float16* x_h    = (_Float16*)carve((size_t)NNODES * DPAD * 2);
  _Float16* Win_h  = (_Float16*)carve((size_t)H0 * DPAD * 2);
  _Float16* Wl1_h  = (_Float16*)carve((size_t)H1 * H0 * 2);
  _Float16* Wr1_h  = (_Float16*)carve((size_t)H1 * H0 * 2);
  _Float16* Wl2_h  = (_Float16*)carve((size_t)H2 * H1 * 2);
  _Float16* Wr2_h  = (_Float16*)carve((size_t)H2 * H1 * 2);
  _Float16* sim    = (_Float16*)carve((size_t)NNODES * NNODES * 2);  // becomes adj
  _Float16* h0     = (_Float16*)carve((size_t)NNODES * H0 * 2);
  _Float16* h0T    = (_Float16*)carve((size_t)H0 * NNODES * 2);
  _Float16* aggr1  = (_Float16*)carve((size_t)NNODES * H0 * 2);
  _Float16* h1     = (_Float16*)carve((size_t)NNODES * H1 * 2);
  _Float16* h1T    = (_Float16*)carve((size_t)H1 * NNODES * 2);
  _Float16* aggr2  = (_Float16*)carve((size_t)NNODES * H1 * 2);
  _Float16* h2     = (_Float16*)carve((size_t)NNODES * H2 * 2);
  float* acc       = (float*)carve((size_t)NNODES * H1 * 4);
  float* inv_deg   = (float*)carve((size_t)NNODES * 4);
  unsigned* bins   = (unsigned*)carve((size_t)NBINS * 4);
  float* eps       = (float*)carve(4);

  // 1) normalize rows -> xn_h ; plain f16 copies of x and all weights
  rownorm_kernel<<<NNODES, 256, 0, stream>>>(x, xn_h, DIN, DPAD);
  convpad_kernel<<<4096, 256, 0, stream>>>(x, x_h, NNODES, DIN, DPAD);
  convpad_kernel<<<1024, 256, 0, stream>>>(W_in, Win_h, H0, DIN, DPAD);
  convpad_kernel<<<512, 256, 0, stream>>>(Wl1, Wl1_h, H1, H0, H0);
  convpad_kernel<<<512, 256, 0, stream>>>(Wr1, Wr1_h, H1, H0, H0);
  convpad_kernel<<<128, 256, 0, stream>>>(Wl2, Wl2_h, H2, H1, H1);
  convpad_kernel<<<128, 256, 0, stream>>>(Wr2, Wr2_h, H2, H1, H1);

  // 2) sim = |xn xn^T|  (M=N=8192, K=2048)
  wmma_gemm_kernel<<<dim3(NNODES / 128, NNODES / 64), 256, 0, stream>>>(
      xn_h, DPAD, xn_h, DPAD, sim, nullptr, NNODES, nullptr, nullptr, DPAD, F_ABS | F_ST16);

  // 3) 99th-percentile epsilon via histogram, then threshold -> adj + 1/deg
  zero_bins_kernel<<<(NBINS + 255) / 256, 256, 0, stream>>>(bins);
  hist_kernel<<<2048, 256, 0, stream>>>(sim, bins);
  find_eps_kernel<<<1, 64, 0, stream>>>(bins, eps);
  thresh_deg_kernel<<<NNODES, 256, 0, stream>>>(sim, inv_deg, eps);

  // 4) h0 = relu(x W_in^T)   (M=8192, N=512, K=2048)
  wmma_gemm_kernel<<<dim3(NNODES / 128, H0 / 64), 256, 0, stream>>>(
      x_h, DPAD, Win_h, DPAD, h0, nullptr, H0, nullptr, nullptr, DPAD, F_RELU | F_ST16);

  // 5) SAGE layer 1
  transpose_f16_kernel<<<dim3(H0 / 32, NNODES / 32), dim3(32, 8), 0, stream>>>(h0, h0T, NNODES, H0);
  wmma_gemm_kernel<<<dim3(NNODES / 128, H0 / 64), 256, 0, stream>>>(  // aggr1 = (adj h0)/deg
      sim, NNODES, h0T, NNODES, aggr1, nullptr, H0, nullptr, inv_deg, NNODES, F_ST16);
  wmma_gemm_kernel<<<dim3(NNODES / 128, H1 / 64), 256, 0, stream>>>(  // acc = aggr1 Wl1^T
      aggr1, H0, Wl1_h, H0, nullptr, acc, H1, nullptr, nullptr, H0, F_STOREACC);
  wmma_gemm_kernel<<<dim3(NNODES / 128, H1 / 64), 256, 0, stream>>>(  // h1 = relu(h0 Wr1^T + acc + bl1)
      h0, H0, Wr1_h, H0, h1, acc, H1, bl1, nullptr, H0, F_LOADACC | F_RELU | F_ST16);

  // 6) SAGE layer 2
  transpose_f16_kernel<<<dim3(H1 / 32, NNODES / 32), dim3(32, 8), 0, stream>>>(h1, h1T, NNODES, H1);
  wmma_gemm_kernel<<<dim3(NNODES / 128, H1 / 64), 256, 0, stream>>>(  // aggr2 = (adj h1)/deg
      sim, NNODES, h1T, NNODES, aggr2, nullptr, H1, nullptr, inv_deg, NNODES, F_ST16);
  wmma_gemm_kernel<<<dim3(NNODES / 128, H2 / 64), 256, 0, stream>>>(  // acc = aggr2 Wl2^T
      aggr2, H1, Wl2_h, H1, nullptr, acc, H2, nullptr, nullptr, H1, F_STOREACC);
  wmma_gemm_kernel<<<dim3(NNODES / 128, H2 / 64), 256, 0, stream>>>(  // h2 = relu(h1 Wr2^T + acc + bl2)
      h1, H1, Wr2_h, H1, h2, acc, H2, bl2, nullptr, H1, F_LOADACC | F_RELU | F_ST16);

  // 7) out = sigmoid(h2 W_out^T + b_out)
  head_kernel<<<(NNODES + 255) / 256, 256, 0, stream>>>(h2, W_out, b_out, out);
}
